// TANEncoderLayer_7421703488070
// MI455X (gfx1250) — matrix operand order repaired and checked by reading, not compiled, not run on hardware
//
#include <hip/hip_runtime.h>

// ---------------- problem constants ----------------
#define Bq  2
#define Sq  2048
#define Eq  768
#define Hq  12
#define Tq  128
#define Kq  16
#define Fq  3072
#define DHq 64
#define BSq (Bq*Sq)

typedef unsigned short u16;
typedef __bf16 bf16_t;
typedef bf16_t v16bf __attribute__((ext_vector_type(16)));
typedef float  v8f   __attribute__((ext_vector_type(8)));
typedef u16    ush8  __attribute__((ext_vector_type(8)));

enum { EPI_NONE=0, EPI_RELU=1, EPI_GELU=2, EPI_SIG_ACC=3, EPI_DIST=4 };

// ---------------- helpers ----------------
__device__ __forceinline__ u16 f2bfu(float f){
  unsigned u = __float_as_uint(f);
  u += 0x7FFFu + ((u >> 16) & 1u);          // round-to-nearest-even
  return (u16)(u >> 16);
}

union FragCvt { struct { ush8 lo, hi; } u; v16bf v; };

// A fragment 16x32 bf16: 16 contiguous K values per lane
__device__ __forceinline__ v16bf load_fragA(const u16* p){
  FragCvt f; f.u.lo = *(const ush8*)p; f.u.hi = *(const ush8*)(p+8); return f.v;
}
// B fragment 32x16 bf16: 16 contiguous K values per lane
__device__ __forceinline__ v16bf load_fragB(const u16* p){
  FragCvt f; f.u.lo = *(const ush8*)p; f.u.hi = *(const ush8*)(p+8); return f.v;
}
__device__ __forceinline__ v8f wmma_bf16(v16bf a, v16bf b, v8f c){
  return __builtin_amdgcn_wmma_f32_16x16x32_bf16(false, a, false, b, (short)0, c, false, false);
}

// ---------------- elementwise / prep kernels ----------------
__global__ void tan_transpose_bf16(const float* __restrict__ src, u16* __restrict__ dst,
                                   int R, int C){
  size_t n = (size_t)R*C;
  for (size_t i = (size_t)blockIdx.x*blockDim.x + threadIdx.x; i < n;
       i += (size_t)gridDim.x*blockDim.x){
    int r = (int)(i / C), c = (int)(i % C);
    dst[(size_t)c*R + r] = f2bfu(src[i]);
  }
}

__global__ void tan_convert_bf16(const float* __restrict__ src, u16* __restrict__ dst, size_t n){
  for (size_t i = (size_t)blockIdx.x*blockDim.x + threadIdx.x; i < n;
       i += (size_t)gridDim.x*blockDim.x)
    dst[i] = f2bfu(src[i]);
}

__global__ __launch_bounds__(256) void tan_rownorm_bf16(const float* __restrict__ in,
                                                        u16* __restrict__ out, int n){
  __shared__ float red[256];
  int row = blockIdx.x;
  const float* p = in + (size_t)row*n;
  float s = 0.f;
  for (int i = threadIdx.x; i < n; i += 256){ float v = p[i]; s += v*v; }
  red[threadIdx.x] = s; __syncthreads();
  for (int st = 128; st > 0; st >>= 1){
    if (threadIdx.x < st) red[threadIdx.x] += red[threadIdx.x + st];
    __syncthreads();
  }
  float inv = 1.f / (sqrtf(red[0]) + 1e-8f);
  for (int i = threadIdx.x; i < n; i += 256) out[(size_t)row*n + i] = f2bfu(p[i]*inv);
}

__global__ __launch_bounds__(256) void tan_layernorm(const float* __restrict__ in,
    const float* __restrict__ sc, const float* __restrict__ bi,
    float* __restrict__ outf, u16* __restrict__ outb, int n){
  __shared__ float red[256];
  int row = blockIdx.x;
  const float* p = in + (size_t)row*n;
  float s = 0.f;
  for (int i = threadIdx.x; i < n; i += 256) s += p[i];
  red[threadIdx.x] = s; __syncthreads();
  for (int st = 128; st > 0; st >>= 1){
    if (threadIdx.x < st) red[threadIdx.x] += red[threadIdx.x + st];
    __syncthreads();
  }
  float mu = red[0] / n;
  __syncthreads();
  float s2 = 0.f;
  for (int i = threadIdx.x; i < n; i += 256){ float d = p[i]-mu; s2 += d*d; }
  red[threadIdx.x] = s2; __syncthreads();
  for (int st = 128; st > 0; st >>= 1){
    if (threadIdx.x < st) red[threadIdx.x] += red[threadIdx.x + st];
    __syncthreads();
  }
  float rstd = rsqrtf(red[0]/n + 1e-6f);
  for (int i = threadIdx.x; i < n; i += 256){
    float y = (p[i]-mu)*rstd*sc[i] + bi[i];
    if (outf) outf[(size_t)row*n + i] = y;
    if (outb) outb[(size_t)row*n + i] = f2bfu(y);
  }
}

// ---------------- generic bf16 WMMA GEMM, fused epilogues ----------------
// Block tile 128x128, 8 waves as 4(M)x2(N); each wave 32x64:
// per K-step 2 A-frags + 4 B-frags feed 8 WMMAs (A reused x4, B reused x2).
__global__ __launch_bounds__(256) void tan_gemm_bf16(
    const u16* __restrict__ A, const u16* __restrict__ Bt,
    const float* __restrict__ bias, const float* __restrict__ addsrc,
    float* __restrict__ C, u16* __restrict__ Cb,
    int M, int N, int K, int epi){
  int lane = threadIdx.x & 31;
  int w    = threadIdx.x >> 5;
  int wm = w >> 1, wn = w & 1;                   // 4 (M) x 2 (N) waves
  int rowBase = blockIdx.y*128 + wm*32;
  int colBase = blockIdx.x*128 + wn*64;
  int m  = lane & 15;
  int kA = (lane >> 4) << 4;
  int kB = (lane >> 4) << 4;
  v8f acc[2][4];
  #pragma unroll
  for (int i = 0; i < 2; ++i)
    #pragma unroll
    for (int j = 0; j < 4; ++j) acc[i][j] = (v8f){};
  const u16* pA0 = A  + (size_t)(rowBase + m)*K + kA;
  const u16* pA1 = pA0 + (size_t)16*K;
  const u16* pB0 = Bt + (size_t)(colBase + m)*K + kB;
  for (int kt = 0; kt < K; kt += 32){
    __builtin_prefetch(pA0 + kt + 256, 0, 0);     // speculative next-tile prefetch
    __builtin_prefetch(pB0 + kt + 256, 0, 0);
    v16bf a0 = load_fragA(pA0 + kt);
    v16bf a1 = load_fragA(pA1 + kt);
    #pragma unroll
    for (int j = 0; j < 4; ++j){
      v16bf b = load_fragB(pB0 + (size_t)(j*16)*K + kt);
      acc[0][j] = wmma_bf16(a0, b, acc[0][j]);
      acc[1][j] = wmma_bf16(a1, b, acc[1][j]);
    }
  }
  int rO = (lane >> 4) << 3;
  #pragma unroll
  for (int im = 0; im < 2; ++im){
    #pragma unroll
    for (int jn = 0; jn < 4; ++jn){
      #pragma unroll
      for (int vv = 0; vv < 8; ++vv){
        int r = rowBase + im*16 + rO + vv;
        int c = colBase + jn*16 + m;
        float x = acc[im][jn][vv];
        if (bias) x += bias[c];
        size_t o = (size_t)r*N + c;
        if (epi == EPI_RELU)      x = fmaxf(x, 0.f);
        else if (epi == EPI_GELU){
          float x3 = x*x*x;
          x = 0.5f*x*(1.f + tanhf(0.7978845608028654f*(x + 0.044715f*x3)));
        } else if (epi == EPI_SIG_ACC){
          x += C[o]; x = 1.f/(1.f + expf(-x));
        } else if (epi == EPI_DIST){
          x = 1.f - x; if (r == c) x = 1e9f;
        }
        if (addsrc) x += addsrc[o];
        C[o] = x;
        if (Cb) Cb[o] = f2bfu(x);
      }
    }
  }
}

// ---------------- top-16 smallest (stable, ascending) ----------------
__global__ __launch_bounds__(256) void tan_topk16(const float* __restrict__ dist,
                                                  int* __restrict__ idx, float* __restrict__ nd){
  int row = blockIdx.x*blockDim.x + threadIdx.x;
  if (row >= BSq) return;
  int b = row / Sq, s = row % Sq;
  const float* d = dist + (size_t)b*Sq*Sq + (size_t)s*Sq;
  float v[17]; int id[17];
  #pragma unroll
  for (int k = 0; k < 17; ++k){ v[k] = 3.4e38f; id[k] = 0x7fffffff; }
  for (int j = 0; j < Sq; ++j){
    float dv = d[j];
    if (dv < v[15] || (dv == v[15] && j < id[15])){
      v[16] = dv; id[16] = j;
      #pragma unroll
      for (int p = 16; p >= 1; --p){
        bool sw = (v[p] < v[p-1]) || (v[p] == v[p-1] && id[p] < id[p-1]);
        if (sw){ float tv=v[p]; v[p]=v[p-1]; v[p-1]=tv; int ti=id[p]; id[p]=id[p-1]; id[p-1]=ti; }
      }
    }
  }
  #pragma unroll
  for (int k = 0; k < 16; ++k){
    idx[(size_t)row*16 + k] = id[k];
    nd [(size_t)row*16 + k] = v[k];
  }
}

// ---------------- kNN softmax combine ----------------
__global__ __launch_bounds__(128) void tan_combine(const float* __restrict__ topo,
    const int* __restrict__ idx, const float* __restrict__ nd, u16* __restrict__ combb){
  __shared__ float w[16];
  int row = blockIdx.x;
  int b = row / Sq;
  if (threadIdx.x == 0){
    float tw[16];
    float mx = -3.4e38f;
    #pragma unroll
    for (int k = 0; k < 16; ++k) mx = fmaxf(mx, -nd[(size_t)row*16 + k]);
    float sum = 0.f;
    #pragma unroll
    for (int k = 0; k < 16; ++k){ tw[k] = expf(-nd[(size_t)row*16 + k] - mx); sum += tw[k]; }
    #pragma unroll
    for (int k = 0; k < 16; ++k) w[k] = tw[k]/sum;
  }
  __syncthreads();
  int t = threadIdx.x;
  float acc = topo[(size_t)row*Tq + t];
  #pragma unroll
  for (int k = 0; k < 16; ++k){
    int nb = b*Sq + idx[(size_t)row*16 + k];
    acc += w[k] * topo[(size_t)nb*Tq + t];
  }
  combb[(size_t)row*Tq + t] = f2bfu(acc);
}

// ---------------- head reshape (q pre-scaled, V transposed) ----------------
__global__ void tan_qkv_reshape(const float* __restrict__ q, const float* __restrict__ k,
    const float* __restrict__ v, u16* __restrict__ qh, u16* __restrict__ kh,
    u16* __restrict__ vT){
  size_t n = (size_t)BSq*Eq;
  for (size_t i = (size_t)blockIdx.x*blockDim.x + threadIdx.x; i < n;
       i += (size_t)gridDim.x*blockDim.x){
    int sg = (int)(i / Eq), col = (int)(i % Eq);
    int b = sg / Sq, s = sg % Sq;
    int hh = col / DHq, d = col % DHq;
    size_t ho = (size_t)(b*Hq + hh);
    qh[(ho*Sq + s)*DHq + d] = f2bfu(q[i] * 0.125f);   // 1/sqrt(DH)
    kh[(ho*Sq + s)*DHq + d] = f2bfu(k[i]);
    vT[(ho*DHq + d)*Sq + s] = f2bfu(v[i]);
  }
}

// ---------------- flash attention (WMMA QK^T and PV, streaming softmax) ----
__global__ __launch_bounds__(256) void tan_flash_attn(
    const u16* __restrict__ qh, const u16* __restrict__ kh, const u16* __restrict__ vT,
    const int* __restrict__ mask, float* __restrict__ ctx){
  __shared__ float sS[64][68];
  __shared__ u16   sP[64][72];
  __shared__ float sM[64], sL[64], sC[64];
  int tid = threadIdx.x, lane = tid & 31, w = tid >> 5;
  int wm = w & 3, wn = w >> 2;
  int bh = blockIdx.y; int b = bh / Hq, hh = bh % Hq;
  int q0 = blockIdx.x * 64;
  const u16* qp = qh + (size_t)bh*Sq*DHq;
  const u16* kp = kh + (size_t)bh*Sq*DHq;
  const u16* vp = vT + (size_t)bh*DHq*Sq;
  int m = lane & 15, kA = (lane >> 4) << 4, kB = (lane >> 4) << 4;
  if (tid < 64){ sM[tid] = -1e30f; sL[tid] = 0.f; }
  const u16* qrow = qp + (size_t)(q0 + wm*16 + m)*DHq;
  v16bf qa0 = load_fragA(qrow + 0  + kA);
  v16bf qa1 = load_fragA(qrow + 32 + kA);
  v8f o0 = {}, o1 = {};
  int rO = (lane >> 4) << 3;
  __syncthreads();
  for (int j0 = 0; j0 < Sq; j0 += 64){
    // ---- scores S = Q K^T (64x64), per-wave 16x32 ----
    v8f s0 = {}, s1 = {};
    {
      const u16* kb0 = kp + (size_t)(j0 + wn*32 + m)*DHq + kB;
      const u16* kb1 = kb0 + (size_t)16*DHq;
      v16bf b0 = load_fragB(kb0);      v16bf b1 = load_fragB(kb1);
      s0 = wmma_bf16(qa0, b0, s0);     s1 = wmma_bf16(qa0, b1, s1);
      b0 = load_fragB(kb0 + 32);       b1 = load_fragB(kb1 + 32);
      s0 = wmma_bf16(qa1, b0, s0);     s1 = wmma_bf16(qa1, b1, s1);
    }
    #pragma unroll
    for (int vv = 0; vv < 8; ++vv){
      int r = wm*16 + rO + vv;
      int c0 = wn*32 + m, c1 = c0 + 16;
      float x0 = s0[vv], x1 = s1[vv];
      if (mask[b*Sq + j0 + c0] == 0) x0 = -10000.f;
      if (mask[b*Sq + j0 + c1] == 0) x1 = -10000.f;
      sS[r][c0] = x0; sS[r][c1] = x1;
    }
    __syncthreads();
    // ---- streaming softmax update, one thread per query row ----
    if (tid < 64){
      float mo = sM[tid], rmax = mo;
      for (int c = 0; c < 64; ++c) rmax = fmaxf(rmax, sS[tid][c]);
      float corr = expf(mo - rmax);
      float l = sL[tid]*corr;
      for (int c = 0; c < 64; ++c){
        float p = expf(sS[tid][c] - rmax);
        l += p; sP[tid][c] = f2bfu(p);
      }
      sM[tid] = rmax; sL[tid] = l; sC[tid] = corr;
    }
    __syncthreads();
    #pragma unroll
    for (int vv = 0; vv < 8; ++vv){
      float cr = sC[wm*16 + rO + vv];
      o0[vv] *= cr; o1[vv] *= cr;
    }
    // ---- O += P V (P from LDS, V^T from global) ----
    {
      const u16* pr  = &sP[wm*16 + m][0];
      const u16* vb0 = vp + (size_t)(wn*32 + m)*Sq + j0 + kB;
      const u16* vb1 = vb0 + (size_t)16*Sq;
      v16bf a0 = load_fragA(pr + 0  + kA);
      v16bf a1 = load_fragA(pr + 32 + kA);
      v16bf b0 = load_fragB(vb0);      v16bf b1 = load_fragB(vb1);
      o0 = wmma_bf16(a0, b0, o0);      o1 = wmma_bf16(a0, b1, o1);
      b0 = load_fragB(vb0 + 32);       b1 = load_fragB(vb1 + 32);
      o0 = wmma_bf16(a1, b0, o0);      o1 = wmma_bf16(a1, b1, o1);
    }
    __syncthreads();
  }
  #pragma unroll
  for (int vv = 0; vv < 8; ++vv){
    int r = wm*16 + rO + vv;
    float inv = 1.f / sL[r];
    size_t orow = (size_t)(b*Sq + q0 + r)*Eq + hh*DHq;
    ctx[orow + wn*32 + m]      = o0[vv]*inv;
    ctx[orow + wn*32 + 16 + m] = o1[vv]*inv;
  }
}

// ---------------- gated blend ----------------
__global__ void tan_gate_blend(const float* __restrict__ g, const float* __restrict__ c,
                               const float* __restrict__ x, u16* __restrict__ out, size_t n){
  for (size_t i = (size_t)blockIdx.x*blockDim.x + threadIdx.x; i < n;
       i += (size_t)gridDim.x*blockDim.x){
    float gv = g[i];
    out[i] = f2bfu(gv*c[i] + (1.f - gv)*x[i]);
  }
}

// ---------------- host orchestration ----------------
extern "C" void kernel_launch(void* const* d_in, const int* in_sizes, int n_in,
                              void* d_out, int out_size, void* d_ws, size_t ws_size,
                              hipStream_t stream){
  (void)in_sizes; (void)n_in; (void)out_size; (void)ws_size;
  const float* hid    = (const float*)d_in[0];
  const int*   mask   = (const int*)  d_in[1];
  const float* W_topo = (const float*)d_in[2];  const float* b_topo = (const float*)d_in[3];
  const float* W_m1   = (const float*)d_in[4];  const float* b_m1   = (const float*)d_in[5];
  const float* W_m2   = (const float*)d_in[6];  const float* b_m2   = (const float*)d_in[7];
  const float* ln_t_s = (const float*)d_in[8];  const float* ln_t_b = (const float*)d_in[9];
  const float* W_p1   = (const float*)d_in[10]; const float* b_p1   = (const float*)d_in[11];
  const float* W_p2   = (const float*)d_in[12]; const float* b_p2   = (const float*)d_in[13];
  const float* ln1_s  = (const float*)d_in[14]; const float* ln1_b  = (const float*)d_in[15];
  const float* Wq     = (const float*)d_in[16]; const float* Wk     = (const float*)d_in[17];
  const float* Wv     = (const float*)d_in[18];
  const float* W_gate = (const float*)d_in[19]; const float* b_gate = (const float*)d_in[20];
  const float* W_o    = (const float*)d_in[21]; const float* b_o    = (const float*)d_in[22];
  const float* ln2_s  = (const float*)d_in[23]; const float* ln2_b  = (const float*)d_in[24];
  const float* W_f1   = (const float*)d_in[25]; const float* b_f1   = (const float*)d_in[26];
  const float* W_f2   = (const float*)d_in[27]; const float* b_f2   = (const float*)d_in[28];
  float* out = (float*)d_out;

  // ---- workspace bump allocator (256B aligned) ----
  char* ws = (char*)d_ws;
  size_t off = 0;
  auto alloc = [&](size_t bytes) -> void* {
    void* p = ws + off;
    off += (bytes + 255) & ~(size_t)255;
    return p;
  };
  const size_t BSE = (size_t)BSq*Eq, BST = (size_t)BSq*Tq;

  // bf16 transposed weights [N][K]
  u16* w_topoT = (u16*)alloc((size_t)Tq*Eq*2);
  u16* w_m1T   = (u16*)alloc((size_t)2*Tq*Tq*2);
  u16* w_m2T   = (u16*)alloc((size_t)Tq*2*Tq*2);
  u16* w_p1T   = (u16*)alloc((size_t)Tq*Tq*2);
  u16* w_p2T   = (u16*)alloc((size_t)Tq*Tq*2);
  u16* w_qT    = (u16*)alloc((size_t)Eq*Eq*2);
  u16* w_kT    = (u16*)alloc((size_t)Eq*Eq*2);
  u16* w_vT    = (u16*)alloc((size_t)Eq*Eq*2);
  u16* w_g1T   = (u16*)alloc((size_t)Eq*Eq*2);
  u16* w_g2T   = (u16*)alloc((size_t)Eq*Tq*2);
  u16* w_oT    = (u16*)alloc((size_t)Eq*Eq*2);
  u16* w_f1T   = (u16*)alloc((size_t)Fq*Eq*2);
  u16* w_f2T   = (u16*)alloc((size_t)Eq*Fq*2);
  // f32 activations
  float* topo  = (float*)alloc(BST*4);
  float* tmp1  = (float*)alloc((size_t)BSq*2*Tq*4);
  float* tvec  = (float*)alloc(BST*4);
  float* tmp2  = (float*)alloc(BST*4);
  float* pff   = (float*)alloc(BST*4);
  float* xf    = (float*)alloc(BSE*4);
  float* gatef = (float*)alloc(BSE*4);
  float* ctxf  = (float*)alloc(BSE*4);
  float* h2    = (float*)alloc(BSE*4);
  float* ndb   = (float*)alloc((size_t)BSq*Kq*4);
  int*   idxb  = (int*)  alloc((size_t)BSq*Kq*4);
  // bf16 activations
  u16* hb     = (u16*)alloc(BSE*2);
  u16* hnb    = (u16*)alloc(BSE*2);
  u16* xb     = (u16*)alloc(BSE*2);
  u16* combb  = (u16*)alloc(BST*2);
  u16* tmp1b  = (u16*)alloc((size_t)BSq*2*Tq*2);
  u16* tlnb   = (u16*)alloc(BST*2);
  u16* tmp2b  = (u16*)alloc(BST*2);
  u16* pfb    = (u16*)alloc(BST*2);
  u16* ctxb   = (u16*)alloc(BSE*2);
  u16* ctx2b  = (u16*)alloc(BSE*2);
  u16* yb     = (u16*)alloc(BSE*2);
  u16* f1b    = (u16*)alloc((size_t)BSq*Fq*2);
  u16* qhb    = (u16*)alloc(BSE*2);
  u16* khb    = (u16*)alloc(BSE*2);
  u16* vTb    = (u16*)alloc(BSE*2);
  // aliased big region: dist (33.5MB) -> qkv f32 (37.7MB) -> ffn-mid f32 (50.3MB)
  float* U1   = (float*)alloc((size_t)BSq*Fq*4);
  float* dist = U1;
  float* qf = U1; float* kf = U1 + BSE; float* vf = U1 + 2*BSE;
  float* f1f = U1;

  const int EB = 512;  // grid-stride elementwise blocks
  auto tp = [&](const float* s, u16* d, int R, int C){
    tan_transpose_bf16<<<EB, 256, 0, stream>>>(s, d, R, C);
  };
  auto gemm = [&](const u16* A, const u16* Bt, const float* bias, const float* add,
                  float* C, u16* Cb, int M, int N, int K, int epi){
    tan_gemm_bf16<<<dim3(N/128, M/128), 256, 0, stream>>>(A, Bt, bias, add, C, Cb, M, N, K, epi);
  };

  // 1. weight prep (transpose + bf16)
  tp(W_topo, w_topoT, Eq, Tq);
  tp(W_m1, w_m1T, Tq, 2*Tq);
  tp(W_m2, w_m2T, 2*Tq, Tq);
  tp(W_p1, w_p1T, Tq, Tq);
  tp(W_p2, w_p2T, Tq, Tq);
  tp(Wq, w_qT, Eq, Eq); tp(Wk, w_kT, Eq, Eq); tp(Wv, w_vT, Eq, Eq);
  tp(W_gate, w_g1T, Eq, Eq);
  tp(W_gate + (size_t)Eq*Eq, w_g2T, Tq, Eq);
  tp(W_o, w_oT, Eq, Eq);
  tp(W_f1, w_f1T, Eq, Fq);
  tp(W_f2, w_f2T, Fq, Eq);

  // 2. topo = h @ W_topo + b_topo
  tan_convert_bf16<<<EB, 256, 0, stream>>>(hid, hb, BSE);
  gemm(hb, w_topoT, b_topo, nullptr, topo, nullptr, BSq, Tq, Eq, EPI_NONE);

  // 3. cosine distance Gram matrix, fused 1-x / diag epilogue
  tan_rownorm_bf16<<<BSq, 256, 0, stream>>>(hid, hnb, Eq);
  for (int b = 0; b < Bq; ++b)
    gemm(hnb + (size_t)b*Sq*Eq, hnb + (size_t)b*Sq*Eq, nullptr, nullptr,
         dist + (size_t)b*Sq*Sq, nullptr, Sq, Sq, Eq, EPI_DIST);

  // 4. kNN top-16 + softmax combine
  tan_topk16<<<BSq/256, 256, 0, stream>>>(dist, idxb, ndb);
  tan_combine<<<BSq, 128, 0, stream>>>(topo, idxb, ndb, combb);

  // 5. topo MLP + LN + persistence MLP
  gemm(combb, w_m1T, b_m1, nullptr, tmp1, tmp1b, BSq, 2*Tq, Tq, EPI_RELU);
  gemm(tmp1b, w_m2T, b_m2, nullptr, tvec, nullptr, BSq, Tq, 2*Tq, EPI_NONE);
  tan_layernorm<<<BSq, 256, 0, stream>>>(tvec, ln_t_s, ln_t_b, nullptr, tlnb, Tq);
  gemm(tlnb, w_p1T, b_p1, nullptr, tmp2, tmp2b, BSq, Tq, Tq, EPI_RELU);
  gemm(tmp2b, w_p2T, b_p2, nullptr, pff, pfb, BSq, Tq, Tq, EPI_NONE);

  // 6. attention: LN -> QKV -> flash
  tan_layernorm<<<BSq, 256, 0, stream>>>(hid, ln1_s, ln1_b, xf, xb, Eq);
  gemm(xb, w_qT, nullptr, nullptr, qf, nullptr, BSq, Eq, Eq, EPI_NONE);
  gemm(xb, w_kT, nullptr, nullptr, kf, nullptr, BSq, Eq, Eq, EPI_NONE);
  gemm(xb, w_vT, nullptr, nullptr, vf, nullptr, BSq, Eq, Eq, EPI_NONE);
  tan_qkv_reshape<<<EB, 256, 0, stream>>>(qf, kf, vf, qhb, khb, vTb);
  tan_flash_attn<<<dim3(Sq/64, Bq*Hq), 256, 0, stream>>>(qhb, khb, vTb, mask, ctxf);

  // 7. gate = sigmoid(ctx@Wg1 + pf@Wg2 + b_gate); blend; out-proj residual
  tan_convert_bf16<<<EB, 256, 0, stream>>>(ctxf, ctxb, BSE);
  gemm(ctxb, w_g1T, b_gate, nullptr, gatef, nullptr, BSq, Eq, Eq, EPI_NONE);
  gemm(pfb,  w_g2T, nullptr, nullptr, gatef, nullptr, BSq, Eq, Tq, EPI_SIG_ACC);
  tan_gate_blend<<<EB, 256, 0, stream>>>(gatef, ctxf, xf, ctx2b, BSE);
  gemm(ctx2b, w_oT, b_o, hid, h2, nullptr, BSq, Eq, Eq, EPI_NONE);

  // 8. FFN block (GELU fused, bf16 side-output), final residual into d_out
  tan_layernorm<<<BSq, 256, 0, stream>>>(h2, ln2_s, ln2_b, nullptr, yb, Eq);
  gemm(yb, w_f1T, b_f1, nullptr, f1f, f1b, BSq, Fq, Eq, EPI_GELU);
  gemm(f1b, w_f2T, b_f2, h2, out, nullptr, BSq, Eq, Fq, EPI_NONE);
}
